// MultiHeadAttention_58368605553017
// MI455X (gfx1250) — matrix operand order repaired and checked by reading, not compiled, not run on hardware
//
#include <hip/hip_runtime.h>
#include <cstdint>

// Problem constants (match reference)
#define BB 4
#define TT 2048
#define EE 1024
#define HH 16
#define HSZ 64
#define BT (BB*TT)   // 8192 rows

typedef __bf16 bf16_t;
typedef __attribute__((ext_vector_type(16))) __bf16 bf16x16;
typedef __attribute__((ext_vector_type(8)))  float  f32x8;

union Frag {
  bf16x16 v;
  uint4   u[2];
};

__device__ __forceinline__ f32x8 wmma_bf16(const Frag& a, const Frag& b, f32x8 c) {
  // D = A(16x32 bf16) * B(32x16 bf16) + C(16x16 f32)
  return __builtin_amdgcn_wmma_f32_16x16x32_bf16(
      /*neg_a=*/false, a.v, /*neg_b=*/false, b.v,
      /*c_mod=*/(short)0, c, /*reuse_a=*/false, /*reuse_b=*/false);
}

// -------- CDNA5 async global->LDS copy (GLOBAL_LOAD_ASYNC_TO_LDS_B128) --------
// Per the compiler diagnostic, the builtin takes GCC-vector int4 pointers:
//   arg0: addrspace(1) v4i*  (global source, prints as "int __device__ *")
//   arg1: addrspace(3) v4i*  (LDS destination)
typedef int v4i __attribute__((vector_size(16)));
typedef __attribute__((address_space(1))) v4i gv4i;
typedef __attribute__((address_space(3))) v4i lv4i;

#if defined(__has_builtin)
#if __has_builtin(__builtin_amdgcn_global_load_async_to_lds_b128)
#define HAVE_ASYNC_LDS 1
#endif
#endif

__device__ __forceinline__ void async_copy16(const void* gsrc, void* ldst) {
#if defined(HAVE_ASYNC_LDS)
  __builtin_amdgcn_global_load_async_to_lds_b128(
      (gv4i*)gsrc, (lv4i*)ldst, 0, 0);
#else
  *(uint4*)ldst = *(const uint4*)gsrc;
#endif
}

__device__ __forceinline__ void async_wait0() {
#if defined(HAVE_ASYNC_LDS)
#if __has_builtin(__builtin_amdgcn_s_wait_asynccnt)
  __builtin_amdgcn_s_wait_asynccnt(0);
#else
  asm volatile("s_wait_asynccnt 0x0" ::: "memory");
#endif
#endif
}

// ---------------- f32 -> bf16 conversion ----------------
__global__ void cvt_bf16_kernel(const float* __restrict__ src,
                                bf16_t* __restrict__ dst, int n) {
  int i = blockIdx.x * blockDim.x + threadIdx.x;
  int stride = gridDim.x * blockDim.x;
  for (; i < n; i += stride) dst[i] = (bf16_t)src[i];
}

// ---------------- generic WMMA GEMM ----------------
// C[M=8192][N=1024] = A[M][K=1024] (bf16, row-major) * W[N][K]^T (bf16, torch layout)
// mode 0: dst = bf16 [B,H,T,HSZ]   (q / k projections)
// mode 2: dst = bf16 [B,H,HSZ,T]   (v projection, transposed for PV matmul)
// mode 3: dst = f32  [M][N] + bias (final output projection)
__global__ __launch_bounds__(128)
void gemm_kernel(const bf16_t* __restrict__ A, const bf16_t* __restrict__ W,
                 void* __restrict__ dst, const float* __restrict__ bias, int mode) {
  __shared__ __align__(16) bf16_t As[2][128][32];
  __shared__ __align__(16) bf16_t Bs[2][64][32];

  const int tid  = threadIdx.x;
  const int wave = tid >> 5, lane = tid & 31;
  const int half = lane >> 4, lq = lane & 15;
  const int wm = wave >> 1, wn = wave & 1;   // 2x2 wave grid
  const int m0 = blockIdx.y * 128;
  const int n0 = blockIdx.x * 64;

  const f32x8 zero = {0.f,0.f,0.f,0.f,0.f,0.f,0.f,0.f};
  f32x8 acc[4][2];
#pragma unroll
  for (int i = 0; i < 4; ++i)
#pragma unroll
    for (int j = 0; j < 2; ++j) acc[i][j] = zero;

  auto stage_tiles = [&](int buf, int k0) {
    // A tile: 128x32 bf16 = 512 x 16B chunks, 4 per thread
#pragma unroll
    for (int i = 0; i < 4; ++i) {
      int c = i * 128 + tid;
      int r = c >> 2, cc = (c & 3) * 8;
      async_copy16(&A[(size_t)(m0 + r) * EE + k0 + cc], &As[buf][r][cc]);
    }
    // W tile (already B^T layout): 64x32 bf16 = 256 chunks, 2 per thread
#pragma unroll
    for (int i = 0; i < 2; ++i) {
      int c = i * 128 + tid;
      int r = c >> 2, cc = (c & 3) * 8;
      async_copy16(&W[(size_t)(n0 + r) * EE + k0 + cc], &Bs[buf][r][cc]);
    }
  };

  stage_tiles(0, 0);
  async_wait0();
  __syncthreads();

  int cur = 0;
  for (int k0 = 0; k0 < EE; k0 += 32) {
    if (k0 + 32 < EE) stage_tiles(cur ^ 1, k0 + 32);  // prefetch next K-slab

    Frag bfrag[2];
#pragma unroll
    for (int j = 0; j < 2; ++j) {
      const bf16_t* p = &Bs[cur][wn * 32 + j * 16 + lq][half * 8];
      bfrag[j].u[0] = *(const uint4*)p;
      bfrag[j].u[1] = *(const uint4*)(p + 16);
    }
#pragma unroll
    for (int i = 0; i < 4; ++i) {
      Frag a;
      const bf16_t* p = &As[cur][wm * 64 + i * 16 + lq][half * 8];
      a.u[0] = *(const uint4*)p;
      a.u[1] = *(const uint4*)(p + 16);
#pragma unroll
      for (int j = 0; j < 2; ++j)
        acc[i][j] = wmma_bf16(a, bfrag[j], acc[i][j]);
    }

    async_wait0();
    __syncthreads();
    cur ^= 1;
  }

  // epilogue: C layout row = half*8+r, col = lq
#pragma unroll
  for (int i = 0; i < 4; ++i) {
#pragma unroll
    for (int j = 0; j < 2; ++j) {
#pragma unroll
      for (int r = 0; r < 8; ++r) {
        int m = m0 + wm * 64 + i * 16 + half * 8 + r;
        int n = n0 + wn * 32 + j * 16 + lq;
        float v = acc[i][j][r];
        if (mode == 3) {
          ((float*)dst)[(size_t)m * EE + n] = v + bias[n];
        } else {
          int b_ = m / TT, t = m % TT;
          int h = n >> 6, d = n & 63;
          size_t idx;
          if (mode == 2) idx = ((size_t)(b_ * HH + h) * HSZ + d) * TT + t;
          else           idx = ((size_t)(b_ * HH + h) * TT + t) * HSZ + d;
          ((bf16_t*)dst)[idx] = (bf16_t)v;
        }
      }
    }
  }
}

// ---------------- flash attention ----------------
// Reference: scores[t,s] = k_t . q_s  -> "query" role is K.
// grid = (T/64, B*H); 128 threads = 4 waves; each wave owns 16 rows t.
__global__ __launch_bounds__(128)
void flash_kernel(const bf16_t* __restrict__ Kb,   // [B,H,T,HSZ]
                  const bf16_t* __restrict__ Qb,   // [B,H,T,HSZ]
                  const bf16_t* __restrict__ Vt,   // [B,H,HSZ,T]
                  bf16_t* __restrict__ Ob) {       // [B,T,E]
  __shared__ __align__(16) bf16_t qs[2][64][64];   // q rows (s-block), double buffered
  __shared__ __align__(16) bf16_t vs[2][64][64];   // Vt tile [d][s], double buffered
  __shared__ __align__(16) bf16_t ps[4][16][64];   // per-wave P staging

  const int tid  = threadIdx.x;
  const int wave = tid >> 5, lane = tid & 31;
  const int half = lane >> 4, lq = lane & 15;
  const int blk  = blockIdx.x;
  const int bh   = blockIdx.y;
  const size_t head = (size_t)bh;
  const int t0w  = blk * 64 + wave * 16;
  const int hcol = (bh % HH) * HSZ;
  const int bidx = bh / HH;

  // preload this wave's 16 K-rows as A fragments (two 32-wide K chunks over HSZ)
  Frag ka[2];
  {
    const bf16_t* kp = Kb + (head * TT + (size_t)(t0w + lq)) * HSZ;
#pragma unroll
    for (int c = 0; c < 2; ++c) {
      int d0 = c * 32 + half * 8;
      ka[c].u[0] = *(const uint4*)(kp + d0);
      ka[c].u[1] = *(const uint4*)(kp + d0 + 16);
    }
  }

  auto stage_sblock = [&](int buf, int s0) {
    // 512 x 16B chunks each for qs and vs
#pragma unroll
    for (int i = 0; i < 4; ++i) {
      int c = i * 128 + tid;
      int r = c >> 3, cc = (c & 7) * 8;
      async_copy16(&Qb[(head * TT + s0 + r) * HSZ + cc], &qs[buf][r][cc]);
      async_copy16(&Vt[(head * HSZ + r) * TT + s0 + cc], &vs[buf][r][cc]);
    }
  };

  const f32x8 zero = {0.f,0.f,0.f,0.f,0.f,0.f,0.f,0.f};
  f32x8 O[4] = {zero, zero, zero, zero};
  float mrow[8], lrow[8];
#pragma unroll
  for (int r = 0; r < 8; ++r) { mrow[r] = -1e30f; lrow[r] = 0.f; }

  stage_sblock(0, 0);
  async_wait0();
  __syncthreads();

  int cur = 0;
  for (int sb = 0; sb <= blk; ++sb) {
    const int s0 = sb * 64;
    if (sb < blk) stage_sblock(cur ^ 1, s0 + 64);  // prefetch next s-block

    // S = K_rows x Q^T : 4 n-tiles of 16 cols, K-dim = HSZ in two 32-chunks
    f32x8 S[4];
#pragma unroll
    for (int j = 0; j < 4; ++j) {
      S[j] = zero;
#pragma unroll
      for (int c = 0; c < 2; ++c) {
        Frag qb;
        const bf16_t* p = &qs[cur][j * 16 + lq][c * 32 + half * 8];
        qb.u[0] = *(const uint4*)p;
        qb.u[1] = *(const uint4*)(p + 16);
        S[j] = wmma_bf16(ka[c], qb, S[j]);
      }
    }

    // scale + causal mask (only the diagonal block needs masking)
    const float sc = 0.125f; // HSZ^-0.5
#pragma unroll
    for (int j = 0; j < 4; ++j) {
      int sg = s0 + j * 16 + lq;
#pragma unroll
      for (int r = 0; r < 8; ++r) {
        float v = S[j][r] * sc;
        if (sb == blk && sg > t0w + half * 8 + r) v = -1e30f;
        S[j][r] = v;
      }
    }

    // online softmax: row max across 4 tiles + 16 lanes of the half-wave
    float rmax[8];
#pragma unroll
    for (int r = 0; r < 8; ++r) {
      float v = fmaxf(fmaxf(S[0][r], S[1][r]), fmaxf(S[2][r], S[3][r]));
      v = fmaxf(v, __shfl_xor(v, 1, 32));
      v = fmaxf(v, __shfl_xor(v, 2, 32));
      v = fmaxf(v, __shfl_xor(v, 4, 32));
      v = fmaxf(v, __shfl_xor(v, 8, 32));
      rmax[r] = v;
    }

    float alpha[8], rsum[8];
#pragma unroll
    for (int r = 0; r < 8; ++r) {
      float mn = fmaxf(mrow[r], rmax[r]);
      alpha[r] = __expf(mrow[r] - mn);
      mrow[r] = mn;
      rsum[r] = 0.f;
    }

    // P = exp(S - m); stage bf16 P tile to LDS in row-major for A-frag reload
#pragma unroll
    for (int j = 0; j < 4; ++j) {
#pragma unroll
      for (int r = 0; r < 8; ++r) {
        float p = __expf(S[j][r] - mrow[r]);
        rsum[r] += p;
        ps[wave][half * 8 + r][j * 16 + lq] = (bf16_t)p;
      }
    }
#pragma unroll
    for (int r = 0; r < 8; ++r) {
      float v = rsum[r];
      v += __shfl_xor(v, 1, 32);
      v += __shfl_xor(v, 2, 32);
      v += __shfl_xor(v, 4, 32);
      v += __shfl_xor(v, 8, 32);
      lrow[r] = lrow[r] * alpha[r] + v;
#pragma unroll
      for (int jd = 0; jd < 4; ++jd) O[jd][r] *= alpha[r];
    }

    // O += P(16x64) x V(64x64); B^T[d][s] = vs tile natively
    Frag pa[2];
#pragma unroll
    for (int c = 0; c < 2; ++c) {
      const bf16_t* p = &ps[wave][lq][c * 32 + half * 8];
      pa[c].u[0] = *(const uint4*)p;
      pa[c].u[1] = *(const uint4*)(p + 16);
    }
#pragma unroll
    for (int jd = 0; jd < 4; ++jd) {
#pragma unroll
      for (int c = 0; c < 2; ++c) {
        Frag vb;
        const bf16_t* p = &vs[cur][jd * 16 + lq][c * 32 + half * 8];
        vb.u[0] = *(const uint4*)p;
        vb.u[1] = *(const uint4*)(p + 16);
        O[jd] = wmma_bf16(pa[c], vb, O[jd]);
      }
    }

    async_wait0();
    __syncthreads();
    cur ^= 1;
  }

  // epilogue: O /= l, write bf16 [B,T,E] (head-major concat)
#pragma unroll
  for (int jd = 0; jd < 4; ++jd) {
#pragma unroll
    for (int r = 0; r < 8; ++r) {
      int t = t0w + half * 8 + r;
      int col = hcol + jd * 16 + lq;
      Ob[((size_t)bidx * TT + t) * EE + col] = (bf16_t)(O[jd][r] / lrow[r]);
    }
  }
}

extern "C" void kernel_launch(void* const* d_in, const int* in_sizes, int n_in,
                              void* d_out, int out_size, void* d_ws, size_t ws_size,
                              hipStream_t stream) {
  (void)in_sizes; (void)n_in; (void)out_size; (void)ws_size;
  const float* x  = (const float*)d_in[0];
  const float* Wk = (const float*)d_in[1];
  const float* Wq = (const float*)d_in[2];
  const float* Wv = (const float*)d_in[3];
  const float* Wo = (const float*)d_in[4];
  const float* bo = (const float*)d_in[5];

  char* ws = (char*)d_ws;
  const size_t MB = 1024ull * 1024ull;
  bf16_t* xb  = (bf16_t*)(ws +  0 * MB);  // [BT, E]        16 MB
  bf16_t* wkb = (bf16_t*)(ws + 16 * MB);  // [E, E]          2 MB
  bf16_t* wqb = (bf16_t*)(ws + 18 * MB);
  bf16_t* wvb = (bf16_t*)(ws + 20 * MB);
  bf16_t* wob = (bf16_t*)(ws + 22 * MB);
  bf16_t* qb  = (bf16_t*)(ws + 24 * MB);  // [B,H,T,HSZ]    16 MB
  bf16_t* kb  = (bf16_t*)(ws + 40 * MB);  // [B,H,T,HSZ]    16 MB
  bf16_t* vtb = (bf16_t*)(ws + 56 * MB);  // [B,H,HSZ,T]    16 MB
  bf16_t* ab  = (bf16_t*)(ws + 72 * MB);  // [B,T,E]        16 MB

  cvt_bf16_kernel<<<2048, 256, 0, stream>>>(x,  xb,  BB * TT * EE);
  cvt_bf16_kernel<<< 512, 256, 0, stream>>>(Wk, wkb, HH * HSZ * EE);
  cvt_bf16_kernel<<< 512, 256, 0, stream>>>(Wq, wqb, HH * HSZ * EE);
  cvt_bf16_kernel<<< 512, 256, 0, stream>>>(Wv, wvb, HH * HSZ * EE);
  cvt_bf16_kernel<<< 512, 256, 0, stream>>>(Wo, wob, EE * EE);

  dim3 g(EE / 64, BT / 128);  // (16, 64)
  gemm_kernel<<<g, 128, 0, stream>>>(xb, wqb, qb,  nullptr, 0);
  gemm_kernel<<<g, 128, 0, stream>>>(xb, wkb, kb,  nullptr, 0);
  gemm_kernel<<<g, 128, 0, stream>>>(xb, wvb, vtb, nullptr, 2);

  flash_kernel<<<dim3(TT / 64, BB * HH), 128, 0, stream>>>(kb, qb, vtb, ab);

  gemm_kernel<<<g, 128, 0, stream>>>(ab, wob, d_out, bo, 3);
}
